// BigBirdSparseAttention_41669772705983
// MI455X (gfx1250) — matrix operand order repaired and checked by reading, not compiled, not run on hardware
//
#include <hip/hip_runtime.h>

typedef __attribute__((ext_vector_type(16))) _Float16 v16h;
typedef __attribute__((ext_vector_type(8)))  _Float16 v8h;
typedef __attribute__((ext_vector_type(8)))  float    v8f;
typedef int v4i __attribute__((vector_size(16)));   // matches builtin param pointee

#define BB   4
#define HH   12
#define SS   4096
#define DD   64
#define BSZ  64
#define FB   (SS / BSZ)   // 64 blocks
#define RR   3

// ---- CDNA5 async global->LDS path (guarded; falls back to sync copies) ----
#if __has_builtin(__builtin_amdgcn_global_load_async_to_lds_b128)
#define HAVE_ASYNC_LDS 1
#else
#define HAVE_ASYNC_LDS 0
#endif

#if HAVE_ASYNC_LDS
// generic->AS1 / generic->AS3 casts via integer round-trip:
// global generic pointers are numerically the global address; for LDS, the
// low 32 bits of a generic pointer are the LDS byte offset (aperture rule).
// Builtin signature (from clang diagnostic): (v4i AS1* src, v4i AS3* dst, imm, imm)
#define GLOBAL_AS4(p) ((__attribute__((address_space(1))) v4i*)(unsigned long long)(uintptr_t)(p))
#define LDS_AS4(p)    ((__attribute__((address_space(3))) v4i*)(unsigned)(uintptr_t)(p))
#if __has_builtin(__builtin_amdgcn_s_wait_asynccnt)
#define WAIT_ASYNC0() __builtin_amdgcn_s_wait_asynccnt(0)
#else
#define WAIT_ASYNC0() asm volatile("s_wait_asynccnt 0x0" ::: "memory")
#endif
#else
#define WAIT_ASYNC0() ((void)0)
#endif

// Issue (or synchronously perform) the copy of one 64x64 f32 K block and one
// 64x64 f32 V block into LDS staging. 128 threads x 8 iters x 16B each.
__device__ __forceinline__ void issue_kv_copy(const float* __restrict__ kp,
                                              const float* __restrict__ vp,
                                              float* Kst, float* Vst, int tid) {
#pragma unroll
    for (int i = 0; i < 8; ++i) {
        int idx = i * 512 + tid * 4;
#if HAVE_ASYNC_LDS
        __builtin_amdgcn_global_load_async_to_lds_b128(GLOBAL_AS4(kp + idx),
                                                       LDS_AS4(Kst + idx), 0, 0);
        __builtin_amdgcn_global_load_async_to_lds_b128(GLOBAL_AS4(vp + idx),
                                                       LDS_AS4(Vst + idx), 0, 0);
#else
        *(float4*)(Kst + idx) = *(const float4*)(kp + idx);
        *(float4*)(Vst + idx) = *(const float4*)(vp + idx);
#endif
    }
}

// Build a 16x32 f16 A-fragment for v_wmma_f32_16x16x32_f16 from two
// contiguous 8-half chunks in LDS (contiguous k order, same as the B loads):
// lanes 0-15 hold K=0..15, lanes 16-31 hold K=16..31.
__device__ __forceinline__ v16h make_a(const _Float16* lo, const _Float16* hi) {
    v8h a = *(const v8h*)lo;
    v8h b = *(const v8h*)hi;
    v16h r;
#pragma unroll
    for (int i = 0; i < 8; ++i) { r[i] = a[i]; r[i + 8] = b[i]; }
    return r;
}

__launch_bounds__(128, 1)
__global__ void bigbird_sparse_attn_kernel(const float* __restrict__ q,
                                           const float* __restrict__ k,
                                           const float* __restrict__ v,
                                           const int*   __restrict__ rnd,
                                           float*       __restrict__ out) {
    // LDS: 8+8+8+16+8+0.5 KB core + 32 KB f32 staging = 80.5 KB
    __shared__ _Float16 Qh[BSZ * DD];   // Q block, f16, row-major, pre-scaled
    __shared__ _Float16 Kh[BSZ * DD];   // K block, f16, [key][d]
    __shared__ _Float16 Vt[DD * BSZ];   // V block transposed, f16, [d][key]
    __shared__ float    Sb[BSZ * BSZ];  // score / output staging, f32
    __shared__ _Float16 Pb[BSZ * BSZ];  // probabilities, f16
    __shared__ float    aB[BSZ];        // per-row rescale alpha
    __shared__ float    lB[BSZ];        // per-row softmax denom
    __shared__ float    Kst[BSZ * DD];  // async staging: K block f32
    __shared__ float    Vst[BSZ * DD];  // async staging: V block f32

    const int blk  = blockIdx.x;
    const int qb   = blk % FB;
    const int h    = (blk / FB) % HH;
    const int b    = blk / (FB * HH);

    const int tid   = threadIdx.x;
    const int wave  = tid >> 5;
    const int ln    = tid & 31;
    const int r16   = ln & 15;
    const int half  = ln >> 4;           // 0: lanes 0-15, 1: lanes 16-31
    const int rbase = wave * 16;         // this wave's 16 query rows

    const size_t bh = ((size_t)b * HH + h) * (size_t)SS * DD;

    // ---- key-block list (uniform across workgroup) ----
    int kbl[8];
    int nkb;
    if (qb == 0)            { kbl[0] = 0; kbl[1] = 1; nkb = 2; }
    else if (qb == 1)       { kbl[0] = 0; kbl[1] = 1; kbl[2] = 2; nkb = 3; }
    else if (qb == FB - 2)  { kbl[0] = FB - 3; kbl[1] = FB - 2; kbl[2] = FB - 1; nkb = 3; }
    else if (qb == FB - 1)  { kbl[0] = FB - 2; kbl[1] = FB - 1; nkb = 2; }
    else {
        kbl[0] = qb - 1; kbl[1] = qb; kbl[2] = qb + 1;   // window
        kbl[3] = 0;      kbl[4] = FB - 1;                 // global
        const int* rp = rnd + (h * (FB - 2) + (qb - 2)) * RR;  // rand_attn[h][qb-2]
        kbl[5] = rp[0]; kbl[6] = rp[1]; kbl[7] = rp[2];
        nkb = 8;
    }

    // ---- kick off async copy of the first K/V block, overlap with Q setup ----
    issue_kv_copy(k + bh + (size_t)kbl[0] * BSZ * DD,
                  v + bh + (size_t)kbl[0] * BSZ * DD, Kst, Vst, tid);

    // ---- load + scale + f16-convert Q block ----
    {
        const float* qp = q + bh + (size_t)qb * BSZ * DD;
#pragma unroll
        for (int i = 0; i < 8; ++i) {
            int idx = i * 512 + tid * 4;
            float4 f = *(const float4*)(qp + idx);
            _Float16* d = &Qh[idx];
            d[0] = (_Float16)(f.x * 0.125f);
            d[1] = (_Float16)(f.y * 0.125f);
            d[2] = (_Float16)(f.z * 0.125f);
            d[3] = (_Float16)(f.w * 0.125f);
        }
    }
    __syncthreads();

    // Q A-fragments (d-chunks 0..31 and 32..63), kept in registers
    v16h aq0, aq1;
    {
        const _Float16* base = &Qh[(rbase + r16) * DD];
        aq0 = make_a(base +      16 * half, base + 8 + 16 * half);
        aq1 = make_a(base + 32 + 16 * half, base + 40 + 16 * half);
    }

    v8f  oacc[4] = {};          // 16x64 f32 output accumulator (4 C-tiles)
    float m_r = -1e30f;         // running row max (per lane; duplicated per half)
    float l_r = 0.0f;           // running row sum

    for (int ib = 0; ib < nkb; ++ib) {
        // ---- staging for block ib has landed (per-wave), make it WG-visible ----
        WAIT_ASYNC0();
        __syncthreads();   // staging complete + previous compute done with Kh/Vt

        // ---- convert staging f32 -> Kh (row-major f16) and Vt (transposed f16) ----
#pragma unroll
        for (int i = 0; i < 8; ++i) {
            int idx = i * 512 + tid * 4;
            float4 f = *(const float4*)&Kst[idx];
            _Float16* d = &Kh[idx];
            d[0] = (_Float16)f.x; d[1] = (_Float16)f.y;
            d[2] = (_Float16)f.z; d[3] = (_Float16)f.w;
            float4 g = *(const float4*)&Vst[idx];
            int key = idx >> 6;   // idx / 64
            int dd  = idx & 63;
            Vt[(dd + 0) * BSZ + key] = (_Float16)g.x;
            Vt[(dd + 1) * BSZ + key] = (_Float16)g.y;
            Vt[(dd + 2) * BSZ + key] = (_Float16)g.z;
            Vt[(dd + 3) * BSZ + key] = (_Float16)g.w;
        }
        __syncthreads();   // Kh/Vt ready; staging buffer free for reuse

        // ---- prefetch next block into staging, overlapped with WMMA+softmax ----
        if (ib + 1 < nkb) {
            issue_kv_copy(k + bh + (size_t)kbl[ib + 1] * BSZ * DD,
                          v + bh + (size_t)kbl[ib + 1] * BSZ * DD, Kst, Vst, tid);
        }

        // ---- S = (Q*scale) x Kb^T : four 16x16 N-tiles, K-dim 64 = 2 WMMAs ----
#pragma unroll
        for (int nt = 0; nt < 4; ++nt) {
            const int key = nt * 16 + r16;                 // B column = key
            const _Float16* kb0 = &Kh[key * DD + 16 * half];
            v16h b0 = *(const v16h*)kb0;          // d = 16*half .. +15 (chunk 0)
            v16h b1 = *(const v16h*)(kb0 + 32);   // d chunk 32..63
            v8f c = {};
            c = __builtin_amdgcn_wmma_f32_16x16x32_f16(false, aq0, false, b0,
                                                       (short)0, c, false, false);
            c = __builtin_amdgcn_wmma_f32_16x16x32_f16(false, aq1, false, b1,
                                                       (short)0, c, false, false);
#pragma unroll
            for (int e = 0; e < 8; ++e)
                Sb[(rbase + e + 8 * half) * BSZ + nt * 16 + r16] = c[e];
        }

        // ---- online softmax over this key block's 64 columns ----
        {
            const float* srow = &Sb[(rbase + r16) * BSZ + half * 32];
            float mx = -1e30f;
#pragma unroll
            for (int c = 0; c < 32; ++c) mx = fmaxf(mx, srow[c]);
            mx = fmaxf(mx, __shfl_xor(mx, 16, 32));
            float m_new = fmaxf(m_r, mx);
            float alpha = __expf(m_r - m_new);
            float sum = 0.0f;
            _Float16* prow = &Pb[(rbase + r16) * BSZ + half * 32];
#pragma unroll
            for (int c = 0; c < 32; ++c) {
                float p = __expf(srow[c] - m_new);
                sum += p;
                prow[c] = (_Float16)p;
            }
            sum += __shfl_xor(sum, 16, 32);
            l_r = l_r * alpha + sum;
            m_r = m_new;
            if (half == 0) aB[rbase + r16] = alpha;
        }
        __syncthreads();

        // ---- rescale accumulators by alpha(row) ----
#pragma unroll
        for (int e = 0; e < 8; ++e) {
            float al = aB[rbase + 8 * half + e];
            oacc[0][e] *= al; oacc[1][e] *= al;
            oacc[2][e] *= al; oacc[3][e] *= al;
        }

        // ---- O += P x V : A = P (16x32 keys), B = Vt rows (contiguous keys) ----
        {
            const _Float16* prow = &Pb[(rbase + r16) * BSZ];
            v16h ap0 = make_a(prow +      16 * half, prow + 8 + 16 * half); // keys 0..31
            v16h ap1 = make_a(prow + 32 + 16 * half, prow + 40 + 16 * half); // keys 32..63
#pragma unroll
            for (int nt = 0; nt < 4; ++nt) {
                const int dc = nt * 16 + r16;              // B column = output feature
                const _Float16* vb = &Vt[dc * BSZ + 16 * half];
                v16h b0 = *(const v16h*)vb;          // keys 16*half .. +15
                v16h b1 = *(const v16h*)(vb + 32);   // keys 32 + ...
                oacc[nt] = __builtin_amdgcn_wmma_f32_16x16x32_f16(false, ap0, false, b0,
                                                                  (short)0, oacc[nt], false, false);
                oacc[nt] = __builtin_amdgcn_wmma_f32_16x16x32_f16(false, ap1, false, b1,
                                                                  (short)0, oacc[nt], false, false);
            }
        }
    }

    // ---- epilogue: normalize and write to (B, S, H, D) ----
    if (half == 0) lB[rbase + r16] = l_r;
#pragma unroll
    for (int nt = 0; nt < 4; ++nt)
#pragma unroll
        for (int e = 0; e < 8; ++e)
            Sb[(rbase + e + 8 * half) * BSZ + nt * 16 + r16] = oacc[nt][e];
    __syncthreads();

    const size_t obase = ((size_t)b * SS + (size_t)qb * BSZ) * (size_t)(HH * DD)
                       + (size_t)h * DD;
#pragma unroll
    for (int rr = 0; rr < 16; ++rr) {
        int row = rbase + rr;
        float inv = 1.0f / lB[row];
        float2 val;
        val.x = Sb[row * BSZ + ln * 2 + 0] * inv;
        val.y = Sb[row * BSZ + ln * 2 + 1] * inv;
        *(float2*)(out + obase + (size_t)row * (HH * DD) + ln * 2) = val;
    }
}

extern "C" void kernel_launch(void* const* d_in, const int* in_sizes, int n_in,
                              void* d_out, int out_size, void* d_ws, size_t ws_size,
                              hipStream_t stream) {
    (void)in_sizes; (void)n_in; (void)out_size; (void)d_ws; (void)ws_size;
    const float* q   = (const float*)d_in[0];   // (B,H,S,D) f32
    const float* k   = (const float*)d_in[1];   // (B,H,S,D) f32
    const float* v   = (const float*)d_in[2];   // (B,H,S,D) f32
    const int*   rnd = (const int*)  d_in[8];   // (H, FB-2, R) i32
    float*       out = (float*)d_out;           // (B,S,H,D) f32
    dim3 grid(BB * HH * FB);   // 3072 workgroups, one per (b,h,q-block)
    bigbird_sparse_attn_kernel<<<grid, 128, 0, stream>>>(q, k, v, rnd, out);
}
